// MPNN_76768245449706
// MI455X (gfx1250) — matrix operand order repaired and checked by reading, not compiled, not run on hardware
//
#include <hip/hip_runtime.h>
#include <hip/hip_bf16.h>
#include <cstdint>

// ---------------------------------------------------------------------------
// Problem constants (from reference): B=32, N=512, F=128, E=4, R=256, 4 steps
// ---------------------------------------------------------------------------
#define ROWS   16384   // B*N
#define FEAT_  128
#define EDGES  4
#define REPR_  256
#define NNODE  512

typedef __attribute__((ext_vector_type(16))) __bf16 v16bf;
typedef __attribute__((ext_vector_type(8)))  float  v8f;

// bf16 round-to-nearest-even truncation
__device__ __forceinline__ uint16_t f2bf(float f) {
    uint32_t u = __builtin_bit_cast(uint32_t, f);
    u += 0x7FFFu + ((u >> 16) & 1u);
    return (uint16_t)(u >> 16);
}

__device__ __forceinline__ float sigmoidf_(float x) {
    return 1.0f / (1.0f + __expf(-x));
}
__device__ __forceinline__ float tanhf_(float x) {
    float cx = fminf(fmaxf(x, -15.0f), 15.0f);
    float e = __expf(2.0f * cx);
    return (e - 1.0f) / (e + 1.0f);
}

// ---------------------------------------------------------------------------
// WMMA helpers.  bf16 16x16x32, fp32 accumulate.
//
// A operand (16x32, row-major source, stride lda in elements):
//   lanes 0-15  hold row M=lane,    K = {k..k+7} and {k+16..k+23}
//   lanes 16-31 hold row M=lane-16, K = {k+8..k+15} and {k+24..k+31}
// B operand (32x16): lane holds column N=lane&15; same K split as A
//   (weights stored [out,in] row-major == column-major KxN).
// ---------------------------------------------------------------------------
__device__ __forceinline__ v16bf load_a_bf(const uint16_t* base, int lda, int lane) {
    int m     = lane & 15;
    int khalf = (lane >> 4) ? 8 : 0;
    const uint16_t* p = base + (size_t)m * lda + khalf;
    union { v16bf v; uint4 q[2]; } u;
    u.q[0] = *(const uint4*)(p);
    u.q[1] = *(const uint4*)(p + 16);
    return u.v;
}

__device__ __forceinline__ v16bf load_b_bf(const uint16_t* base, int ldb, int lane) {
    int n    = lane & 15;
    int koff = (lane >> 4) ? 8 : 0;
    const uint16_t* p = base + (size_t)n * ldb + koff;
    union { v16bf v; uint4 q[2]; } u;
    u.q[0] = *(const uint4*)(p);
    u.q[1] = *(const uint4*)(p + 16);
    return u.v;
}

__device__ __forceinline__ v8f wmma_bf(v16bf a, v16bf b, v8f c) {
    return __builtin_amdgcn_wmma_f32_16x16x32_bf16(
        /*neg_a=*/false, a, /*neg_b=*/false, b,
        /*c_mod=*/(short)0, c, /*reuse_a=*/false, /*reuse_b=*/false);
}

// ---------------------------------------------------------------------------
// Kernel: fp32 -> bf16 conversion (weights + feat)
// ---------------------------------------------------------------------------
__global__ void mpnn_cvt_bf16(const float* __restrict__ src,
                              uint16_t* __restrict__ dst, int n) {
    int i = blockIdx.x * blockDim.x + threadIdx.x;
    if (i < n) dst[i] = f2bf(src[i]);
}

// ---------------------------------------------------------------------------
// Kernel: A[b,s,e] = sum_o adj[b,s,o] * etype[b,s,o,e]   (bandwidth bound)
// One wave per (b,s) row.
// ---------------------------------------------------------------------------
__global__ void mpnn_acoef(const float* __restrict__ adj,
                           const float* __restrict__ etype,
                           float* __restrict__ A) {
    int wave = blockIdx.x * (blockDim.x >> 5) + (threadIdx.x >> 5);
    int lane = threadIdx.x & 31;
    const float* ar = adj   + (size_t)wave * NNODE;
    const float* er = etype + (size_t)wave * NNODE * EDGES;
    float s0 = 0.f, s1 = 0.f, s2 = 0.f, s3 = 0.f;
    for (int o = lane; o < NNODE; o += 32) {
        float  a = ar[o];
        float4 e = *(const float4*)(er + (size_t)o * 4);
        s0 = fmaf(a, e.x, s0); s1 = fmaf(a, e.y, s1);
        s2 = fmaf(a, e.z, s2); s3 = fmaf(a, e.w, s3);
    }
#pragma unroll
    for (int off = 16; off > 0; off >>= 1) {
        s0 += __shfl_down(s0, off);
        s1 += __shfl_down(s1, off);
        s2 += __shfl_down(s2, off);
        s3 += __shfl_down(s3, off);
    }
    if (lane == 0) *(float4*)(A + (size_t)wave * 4) = make_float4(s0, s1, s2, s3);
}

// ---------------------------------------------------------------------------
// Kernel: msg[r, i] = sum_e A[r,e] * (state_bf[r,:] @ W[e]^T)[i]   -> bf16
// One wave per 16x16 output tile: 4 f32 accumulators (one per edge type),
// 16 WMMAs, then per-row scale/sum by A.
// ---------------------------------------------------------------------------
__global__ void mpnn_msg(const uint16_t* __restrict__ state_bf,
                         const uint16_t* __restrict__ Wbf,
                         const float* __restrict__ A,
                         uint16_t* __restrict__ msg_bf) {
    int wave = blockIdx.x * (blockDim.x >> 5) + (threadIdx.x >> 5);
    int lane = threadIdx.x & 31;
    int colbase = (wave & 7) * 16;          // F/16 = 8 col tiles
    int rowbase = (wave >> 3) * 16;         // 1024 row tiles

    v8f acc[EDGES] = {};
#pragma unroll
    for (int k = 0; k < FEAT_; k += 32) {
        v16bf a = load_a_bf(state_bf + (size_t)rowbase * FEAT_ + k, FEAT_, lane);
#pragma unroll
        for (int e = 0; e < EDGES; ++e) {
            v16bf b = load_b_bf(Wbf + ((size_t)e * FEAT_ + colbase) * FEAT_ + k, FEAT_, lane);
            acc[e] = wmma_bf(a, b, acc[e]);
        }
    }
    int n  = lane & 15;
    int mh = (lane >> 4) * 8;
#pragma unroll
    for (int v = 0; v < 8; ++v) {
        int row = rowbase + mh + v;
        float4 c = *(const float4*)(A + (size_t)row * 4);
        float val = c.x * acc[0][v] + c.y * acc[1][v] +
                    c.z * acc[2][v] + c.w * acc[3][v];
        msg_bf[(size_t)row * FEAT_ + colbase + n] = f2bf(val);
    }
}

// ---------------------------------------------------------------------------
// Kernel: fused GRU cell.  One wave per 16x16 tile of the new state.
// 6 accumulators: gi_{r,z,n} = msg @ w_ih^T segments, gh_{r,z,n} = state @ w_hh^T.
// 24 WMMAs per wave, then elementwise gating; writes fp32 + bf16 state.
// ---------------------------------------------------------------------------
__global__ void mpnn_gru(const uint16_t* __restrict__ msg_bf,
                         const uint16_t* __restrict__ state_bf,
                         const float*    __restrict__ state_f32,
                         const uint16_t* __restrict__ wih_bf,
                         const uint16_t* __restrict__ whh_bf,
                         const float* __restrict__ b_ih,
                         const float* __restrict__ b_hh,
                         float*    __restrict__ new_f32,
                         uint16_t* __restrict__ new_bf) {
    int wave = blockIdx.x * (blockDim.x >> 5) + (threadIdx.x >> 5);
    int lane = threadIdx.x & 31;
    int colbase = (wave & 7) * 16;
    int rowbase = (wave >> 3) * 16;

    v8f gi[3] = {};
    v8f gh[3] = {};
#pragma unroll
    for (int k = 0; k < FEAT_; k += 32) {
        v16bf am = load_a_bf(msg_bf   + (size_t)rowbase * FEAT_ + k, FEAT_, lane);
        v16bf as = load_a_bf(state_bf + (size_t)rowbase * FEAT_ + k, FEAT_, lane);
#pragma unroll
        for (int g = 0; g < 3; ++g) {
            v16bf bi = load_b_bf(wih_bf + ((size_t)(g * FEAT_ + colbase)) * FEAT_ + k, FEAT_, lane);
            gi[g] = wmma_bf(am, bi, gi[g]);
            v16bf bh = load_b_bf(whh_bf + ((size_t)(g * FEAT_ + colbase)) * FEAT_ + k, FEAT_, lane);
            gh[g] = wmma_bf(as, bh, gh[g]);
        }
    }
    int n   = lane & 15;
    int mh  = (lane >> 4) * 8;
    int col = colbase + n;
    float bir = b_ih[col], biz = b_ih[FEAT_ + col], bin = b_ih[2 * FEAT_ + col];
    float bhr = b_hh[col], bhz = b_hh[FEAT_ + col], bhn = b_hh[2 * FEAT_ + col];
#pragma unroll
    for (int v = 0; v < 8; ++v) {
        int row = rowbase + mh + v;
        float h  = state_f32[(size_t)row * FEAT_ + col];
        float r  = sigmoidf_((gi[0][v] + bir) + (gh[0][v] + bhr));
        float z  = sigmoidf_((gi[1][v] + biz) + (gh[1][v] + bhz));
        float nn = tanhf_((gi[2][v] + bin) + r * (gh[2][v] + bhn));
        float o  = (1.0f - z) * nn + z * h;
        new_f32[(size_t)row * FEAT_ + col] = o;
        new_bf [(size_t)row * FEAT_ + col] = f2bf(o);
    }
}

// ---------------------------------------------------------------------------
// Kernel: out zero-init (d_out is poisoned by the harness)
// ---------------------------------------------------------------------------
__global__ void mpnn_zero(float* __restrict__ out, int n) {
    int i = blockIdx.x * blockDim.x + threadIdx.x;
    if (i < n) out[i] = 0.0f;
}

// ---------------------------------------------------------------------------
// Kernel: fused readout.
//   att   = sigmoid([feat,state] @ att_w^T + att_b)
//   dense = state @ dense_w^T + dense_b
//   out[b,r] = sum_s dense*att       (atomic fp32 accumulate per 16-row tile)
// One wave per 16x16 tile of (row, R): 12 WMMAs + reduce + 16 atomics.
// ---------------------------------------------------------------------------
__global__ void mpnn_readout(const uint16_t* __restrict__ feat_bf,
                             const uint16_t* __restrict__ state_bf,
                             const uint16_t* __restrict__ attw_bf,
                             const float* __restrict__ att_b,
                             const uint16_t* __restrict__ densew_bf,
                             const float* __restrict__ dense_b,
                             float* __restrict__ out) {
    int wave = blockIdx.x * (blockDim.x >> 5) + (threadIdx.x >> 5);
    int lane = threadIdx.x & 31;
    int colbase = (wave & 15) * 16;         // R/16 = 16 col tiles
    int rowbase = (wave >> 4) * 16;         // 1024 row tiles

    v8f att = {};
    v8f den = {};
#pragma unroll
    for (int k = 0; k < FEAT_; k += 32) {
        v16bf af = load_a_bf(feat_bf  + (size_t)rowbase * FEAT_ + k, FEAT_, lane);
        v16bf as = load_a_bf(state_bf + (size_t)rowbase * FEAT_ + k, FEAT_, lane);
        v16bf bl = load_b_bf(attw_bf + (size_t)colbase * (2 * FEAT_) + k,        2 * FEAT_, lane);
        v16bf br = load_b_bf(attw_bf + (size_t)colbase * (2 * FEAT_) + FEAT_ + k, 2 * FEAT_, lane);
        v16bf bd = load_b_bf(densew_bf + (size_t)colbase * FEAT_ + k, FEAT_, lane);
        att = wmma_bf(af, bl, att);
        att = wmma_bf(as, br, att);
        den = wmma_bf(as, bd, den);
    }
    int n   = lane & 15;
    int col = colbase + n;
    float ab = att_b[col], db = dense_b[col];
    int b = rowbase >> 9;                    // row / 512
    float partial = 0.0f;
#pragma unroll
    for (int v = 0; v < 8; ++v)
        partial += (den[v] + db) * sigmoidf_(att[v] + ab);
    partial += __shfl_down(partial, 16);     // fold lanes 16..31 (rows 8..15)
    if (lane < 16) atomicAdd(&out[(size_t)b * REPR_ + col], partial);
}

// ---------------------------------------------------------------------------
// Host launcher
// ---------------------------------------------------------------------------
extern "C" void kernel_launch(void* const* d_in, const int* in_sizes, int n_in,
                              void* d_out, int out_size, void* d_ws, size_t ws_size,
                              hipStream_t stream) {
    const float* feat    = (const float*)d_in[0];
    const float* etype   = (const float*)d_in[1];
    const float* adj     = (const float*)d_in[2];
    const float* W       = (const float*)d_in[3];
    const float* w_ih    = (const float*)d_in[4];
    const float* w_hh    = (const float*)d_in[5];
    const float* b_ih    = (const float*)d_in[6];
    const float* b_hh    = (const float*)d_in[7];
    const float* att_w   = (const float*)d_in[8];
    const float* att_b   = (const float*)d_in[9];
    const float* dense_w = (const float*)d_in[10];
    const float* dense_b = (const float*)d_in[11];
    float* out = (float*)d_out;

    // ---- workspace carve-out (256 B aligned) -------------------------------
    char* p = (char*)d_ws;
    auto carve = [&](size_t bytes) {
        char* r = p;
        p += (bytes + 255) & ~(size_t)255;
        return r;
    };
    float*    A         = (float*)   carve((size_t)ROWS * EDGES * 4);
    uint16_t* feat_bf   = (uint16_t*)carve((size_t)ROWS * FEAT_ * 2);
    uint16_t* msg_bf    = (uint16_t*)carve((size_t)ROWS * FEAT_ * 2);
    uint16_t* state_bfA = (uint16_t*)carve((size_t)ROWS * FEAT_ * 2);
    uint16_t* state_bfB = (uint16_t*)carve((size_t)ROWS * FEAT_ * 2);
    float*    state_fA  = (float*)   carve((size_t)ROWS * FEAT_ * 4);
    float*    state_fB  = (float*)   carve((size_t)ROWS * FEAT_ * 4);
    uint16_t* W_bf      = (uint16_t*)carve((size_t)EDGES * FEAT_ * FEAT_ * 2);
    uint16_t* wih_bf    = (uint16_t*)carve((size_t)3 * FEAT_ * FEAT_ * 2);
    uint16_t* whh_bf    = (uint16_t*)carve((size_t)3 * FEAT_ * FEAT_ * 2);
    uint16_t* attw_bf   = (uint16_t*)carve((size_t)REPR_ * 2 * FEAT_ * 2);
    uint16_t* densew_bf = (uint16_t*)carve((size_t)REPR_ * FEAT_ * 2);

    auto cvt = [&](const float* s, uint16_t* d, int n) {
        mpnn_cvt_bf16<<<(n + 255) / 256, 256, 0, stream>>>(s, d, n);
    };
    cvt(feat,    feat_bf,   ROWS * FEAT_);
    cvt(W,       W_bf,      EDGES * FEAT_ * FEAT_);
    cvt(w_ih,    wih_bf,    3 * FEAT_ * FEAT_);
    cvt(w_hh,    whh_bf,    3 * FEAT_ * FEAT_);
    cvt(att_w,   attw_bf,   REPR_ * 2 * FEAT_);
    cvt(dense_w, densew_bf, REPR_ * FEAT_);

    // A coefficients: one wave per (b,s) row -> 16384 waves, 4 waves/block
    mpnn_acoef<<<ROWS / 4, 128, 0, stream>>>(adj, etype, A);

    // 4 message-passing steps (ping-pong state buffers; step 0 reads inputs)
    const uint16_t* sb = feat_bf;
    const float*    sf = feat;
    const int gemm_blocks = (ROWS / 16) * (FEAT_ / 16) / 4;   // 8192 waves / 4
    for (int step = 0; step < 4; ++step) {
        mpnn_msg<<<gemm_blocks, 128, 0, stream>>>(sb, W_bf, A, msg_bf);
        float*    nf = (step & 1) ? state_fB  : state_fA;
        uint16_t* nb = (step & 1) ? state_bfB : state_bfA;
        mpnn_gru<<<gemm_blocks, 128, 0, stream>>>(msg_bf, sb, sf, wih_bf, whh_bf,
                                                  b_ih, b_hh, nf, nb);
        sf = nf;
        sb = nb;
    }

    // readout
    mpnn_zero<<<(out_size + 255) / 256, 256, 0, stream>>>(out, out_size);
    const int ro_blocks = (ROWS / 16) * (REPR_ / 16) / 4;     // 16384 waves / 4
    mpnn_readout<<<ro_blocks, 128, 0, stream>>>(feat_bf, sb, attw_bf, att_b,
                                                densew_bf, dense_b, out);
}
